// GINModel_82179904242305
// MI455X (gfx1250) — matrix operand order repaired and checked by reading, not compiled, not run on hardware
//
#include <hip/hip_runtime.h>

#define NN  100000
#define EE  1600000
#define HIDC 128
#define PAD 8
#define LDW (HIDC + PAD)   // 136 halfs per LDS row (272B: staggers banks)

typedef __attribute__((ext_vector_type(16))) _Float16 v16h;
typedef __attribute__((ext_vector_type(8)))  _Float16 v8h;
typedef __attribute__((ext_vector_type(4)))  _Float16 v4h;
typedef __attribute__((ext_vector_type(2)))  _Float16 v2h;
typedef __attribute__((ext_vector_type(8)))  float    v8f;

// hardware packed-f16 atomic add (no-return form, STOREcnt-tracked)
__device__ __forceinline__ void atomic_pk_add_f16(_Float16* p, unsigned int packed2h) {
  asm volatile("global_atomic_pk_add_f16 %0, %1, off"
               :
               : "v"((unsigned long long)(size_t)p), "v"(packed2h)
               : "memory");
}

// ---------------- init: h16 = (f16) x ----------------
__global__ void k_f32_to_f16(const float* __restrict__ x, _Float16* __restrict__ h16, int n4) {
  int i = blockIdx.x * blockDim.x + threadIdx.x;
  if (i < n4) {
    float4 v = ((const float4*)x)[i];
    v4h o;
    o[0] = (_Float16)v.x; o[1] = (_Float16)v.y;
    o[2] = (_Float16)v.z; o[3] = (_Float16)v.w;
    ((v4h*)h16)[i] = o;
  }
}

// ---------------- zero fill (16B granules) ----------------
__global__ void k_zero(float* __restrict__ p, int n4) {
  int i = blockIdx.x * blockDim.x + threadIdx.x;
  if (i < n4) ((float4*)p)[i] = make_float4(0.f, 0.f, 0.f, 0.f);
}

// ---------------- edge scatter: agg16[dst] += h16[src] ----------------
// one wave (32 lanes) per edge; each lane covers 4 consecutive channels
// -> one 8B gather + two global_atomic_pk_add_f16 per lane
__global__ void k_scatter(const _Float16* __restrict__ h16, const int* __restrict__ ei,
                          _Float16* __restrict__ agg16) {
  int gid  = blockIdx.x * blockDim.x + threadIdx.x;
  int e    = gid >> 5;
  int lane = gid & 31;
  if (e >= EE) return;
  int s = ei[e];         // src row
  int d = ei[EE + e];    // dst row
  v4h v = *(const v4h*)(h16 + (size_t)s * HIDC + lane * 4);
  unsigned long long bits = __builtin_bit_cast(unsigned long long, v);
  _Float16* p = agg16 + (size_t)d * HIDC + lane * 4;
  atomic_pk_add_f16(p,     (unsigned int)(bits & 0xffffffffu));
  atomic_pk_add_f16(p + 2, (unsigned int)(bits >> 32));
}

// ---------------- fused GIN MLP layer ----------------
// h16 <- elu( relu( (h16 + agg16) @ wa + ba ) @ wb + bb )   (in place, per 16-row tile)
__global__ __launch_bounds__(256) void k_gin_mlp(
    _Float16* __restrict__ h16, const _Float16* __restrict__ agg16,
    const float* __restrict__ wa, const float* __restrict__ ba,
    const float* __restrict__ wb, const float* __restrict__ bb) {

  __shared__ _Float16 sWA[HIDC * LDW];     // transposed: sWA[n*LDW + k] = wa[k][n]
  __shared__ _Float16 sWB[HIDC * LDW];
  __shared__ _Float16 sT[8 * 16 * LDW];    // per-wave activation tile (X, then H1)
  __shared__ float sBA[HIDC];
  __shared__ float sBB[HIDC];

  const int tid = threadIdx.x;

  // cooperative weight transpose f32->f16 into LDS (coalesced global reads)
  for (int i = tid; i < HIDC * HIDC; i += 256) {
    int k = i >> 7, n = i & 127;
    sWA[n * LDW + k] = (_Float16)wa[i];
    sWB[n * LDW + k] = (_Float16)wb[i];
  }
  if (tid < HIDC) { sBA[tid] = ba[tid]; sBB[tid] = bb[tid]; }
  __syncthreads();

  const int wave = tid >> 5, lane = tid & 31;
  const int tile = blockIdx.x * 8 + wave;
  if (tile * 16 >= NN) return;             // N divisible by 16, last waves idle
  const int row0 = tile * 16;
  _Float16* X = sT + wave * (16 * LDW);

  // stage X = h16 + agg16  (f16), 16x128 tile
  for (int i = lane; i < 512; i += 32) {
    int r = i >> 5, c4 = (i & 31) << 2;
    v4h hv = *(const v4h*)(h16   + (size_t)(row0 + r) * HIDC + c4);
    v4h av = *(const v4h*)(agg16 + (size_t)(row0 + r) * HIDC + c4);
    v4h o;
    o[0] = hv[0] + av[0];
    o[1] = hv[1] + av[1];
    o[2] = hv[2] + av[2];
    o[3] = hv[3] + av[3];
    *(v4h*)(X + r * LDW + c4) = o;
  }

  // lane roles per ISA VGPR layouts
  const int rA   = lane & 15;              // A row (M)
  const int hiHf = (lane >> 4);            // 0: low half lanes, 1: high half lanes
  const int hiA  = hiHf ? 8  : 0;          // A: K base offset within K-block
  const int hiB  = hiHf ? 8  : 0;          // B: K base offset within K-block
  const int outR = hiHf ? 8  : 0;          // C/D: M offset
  const int colB = lane & 15;              // B/D column (N) within tile

  const v8f vzero = {0.f, 0.f, 0.f, 0.f, 0.f, 0.f, 0.f, 0.f};
  v8f acc[8];

  // ---------------- GEMM1: (X @ wa) ----------------
  #pragma unroll
  for (int nt = 0; nt < 8; ++nt) acc[nt] = vzero;

  #pragma unroll
  for (int kb = 0; kb < 4; ++kb) {
    const int k0 = kb * 32;
    v8h alo = *(const v8h*)(X + rA * LDW + k0 + hiA);
    v8h ahi = *(const v8h*)(X + rA * LDW + k0 + hiA + 16);
    v16h A = __builtin_shufflevector(alo, ahi, 0,1,2,3,4,5,6,7,8,9,10,11,12,13,14,15);
    #pragma unroll
    for (int nt = 0; nt < 8; ++nt) {
      const _Float16* wp = sWA + (nt * 16 + colB) * LDW + k0 + hiB;
      v8h blo = *(const v8h*)(wp);
      v8h bhi = *(const v8h*)(wp + 16);
      v16h B = __builtin_shufflevector(blo, bhi, 0,1,2,3,4,5,6,7,8,9,10,11,12,13,14,15);
      acc[nt] = __builtin_amdgcn_wmma_f32_16x16x32_f16(
          false, A, false, B, (short)0, acc[nt], false, false);
    }
  }

  // bias + relu -> write H1 back over X (f16)
  #pragma unroll
  for (int nt = 0; nt < 8; ++nt) {
    float b = sBA[nt * 16 + colB];
    #pragma unroll
    for (int m = 0; m < 8; ++m) {
      float v = acc[nt][m] + b;
      X[(m + outR) * LDW + nt * 16 + colB] = (_Float16)(v > 0.f ? v : 0.f);
    }
  }

  // ---------------- GEMM2: (H1 @ wb) ----------------
  #pragma unroll
  for (int nt = 0; nt < 8; ++nt) acc[nt] = vzero;

  #pragma unroll
  for (int kb = 0; kb < 4; ++kb) {
    const int k0 = kb * 32;
    v8h alo = *(const v8h*)(X + rA * LDW + k0 + hiA);
    v8h ahi = *(const v8h*)(X + rA * LDW + k0 + hiA + 16);
    v16h A = __builtin_shufflevector(alo, ahi, 0,1,2,3,4,5,6,7,8,9,10,11,12,13,14,15);
    #pragma unroll
    for (int nt = 0; nt < 8; ++nt) {
      const _Float16* wp = sWB + (nt * 16 + colB) * LDW + k0 + hiB;
      v8h blo = *(const v8h*)(wp);
      v8h bhi = *(const v8h*)(wp + 16);
      v16h B = __builtin_shufflevector(blo, bhi, 0,1,2,3,4,5,6,7,8,9,10,11,12,13,14,15);
      acc[nt] = __builtin_amdgcn_wmma_f32_16x16x32_f16(
          false, A, false, B, (short)0, acc[nt], false, false);
    }
  }

  // bias + elu -> global (f16, in place)
  #pragma unroll
  for (int nt = 0; nt < 8; ++nt) {
    float b = sBB[nt * 16 + colB];
    #pragma unroll
    for (int m = 0; m < 8; ++m) {
      float v = acc[nt][m] + b;
      v = v > 0.f ? v : (__expf(v) - 1.f);
      h16[(size_t)(row0 + m + outR) * HIDC + nt * 16 + colB] = (_Float16)v;
    }
  }
}

// ---------------- head: out = sigmoid(h @ lin_w + lin_b) ----------------
__global__ void k_final(const _Float16* __restrict__ h16, const float* __restrict__ w,
                        const float* __restrict__ b, float* __restrict__ out) {
  int gid  = blockIdx.x * blockDim.x + threadIdx.x;
  int node = gid >> 5;
  int lane = gid & 31;
  if (node >= NN) return;
  v4h v = *(const v4h*)(h16 + (size_t)node * HIDC + lane * 4);
  float4 wv = *(const float4*)(w + lane * 4);
  float p = (float)v[0] * wv.x + (float)v[1] * wv.y +
            (float)v[2] * wv.z + (float)v[3] * wv.w;
  #pragma unroll
  for (int off = 16; off; off >>= 1) p += __shfl_down(p, off, 32);
  if (lane == 0) out[node] = 1.f / (1.f + __expf(-(p + b[0])));
}

// ---------------- host ----------------
extern "C" void kernel_launch(void* const* d_in, const int* in_sizes, int n_in,
                              void* d_out, int out_size, void* d_ws, size_t ws_size,
                              hipStream_t stream) {
  const float* x  = (const float*)d_in[0];
  const int*   ei = (const int*)d_in[1];
  const float* wA[3] = {(const float*)d_in[2],  (const float*)d_in[6],  (const float*)d_in[10]};
  const float* bA[3] = {(const float*)d_in[3],  (const float*)d_in[7],  (const float*)d_in[11]};
  const float* wB[3] = {(const float*)d_in[4],  (const float*)d_in[8],  (const float*)d_in[12]};
  const float* bB[3] = {(const float*)d_in[5],  (const float*)d_in[9],  (const float*)d_in[13]};
  const float* lw = (const float*)d_in[14];
  const float* lb = (const float*)d_in[15];
  float* out = (float*)d_out;

  // workspace: [0, 25.6MB) f16 activations ; [25.6MB, 51.2MB) f16 agg
  _Float16* h16   = (_Float16*)d_ws;
  _Float16* agg16 = (_Float16*)((char*)d_ws + (size_t)NN * HIDC * sizeof(_Float16));

  const int n4  = NN * HIDC / 4;                // 3.2M float4 quads (f32 convert)
  const int z4  = NN * HIDC * 2 / 16;           // 1.6M 16B granules (f16 agg zero)
  const int cvtBlocks  = (n4 + 255) / 256;
  const int zeroBlocks = (z4 + 255) / 256;
  const int mlpBlocks  = (NN / 16 + 7) / 8;     // 782
  const int scatBlocks = (EE + 7) / 8;          // 200000 (8 edges / 256-thr block)
  const int headBlocks = (NN * 32 + 255) / 256; // wave per node

  k_f32_to_f16<<<cvtBlocks, 256, 0, stream>>>(x, h16, n4);

  for (int l = 0; l < 3; ++l) {
    k_zero<<<zeroBlocks, 256, 0, stream>>>((float*)agg16, z4);
    k_scatter<<<scatBlocks, 256, 0, stream>>>(h16, ei, agg16);
    k_gin_mlp<<<mlpBlocks, 256, 0, stream>>>(h16, agg16, wA[l], bA[l], wB[l], bB[l]);
  }

  k_final<<<headBlocks, 256, 0, stream>>>(h16, lw, lb, out);
}
